// HyperLinear_41205916237909
// MI455X (gfx1250) — compile-verified, matrix operand order checked by reading, not run on hardware
//
#include <hip/hip_runtime.h>

// ---------------------------------------------------------------------------
// HyperLinear on MI455X (gfx1250, wave32, WMMA)
//
// out[t,o] = sum_{i,h} wW3[(o*128+i),h] * h2w[t,h] * x[t,i]      (main GEMM)
//          + sum_i    wb3[o*128+i] * x[t,i]                      (K-extension)
//          + bias[t,o]                                           (C init)
//
// A[t, k=i*32+h] = x[t,i]*h2w[t,h]  (built on the fly, rank-1 per 32-K chunk)
// B[k, o]        = wW3[(o*128+i)*32+h]  (shared, packed once to bf16 frags)
// ---------------------------------------------------------------------------

#define T_TOK   16384   // S*B
#define CIN     128
#define COUT    128
#define HID_    32
#define KCH     128     // main 32-wide K chunks (one per input channel)
#define KCH_TOT 132     // + 4 chunks folding wb3 (K extension by CIN=128)

typedef __attribute__((ext_vector_type(16))) __bf16    v16bf;
typedef __attribute__((ext_vector_type(8)))  float     v8f;
typedef __attribute__((ext_vector_type(8)))  unsigned  v8u;

// round-to-nearest-even f32 -> bf16
__device__ inline unsigned short f2bf(float f) {
  unsigned u = __float_as_uint(f);
  u += 0x7FFFu + ((u >> 16) & 1u);
  return (unsigned short)(u >> 16);
}
__device__ inline unsigned packbf(float lo, float hi) {
  return (unsigned)f2bf(lo) | ((unsigned)f2bf(hi) << 16);
}
// VOP3P packed bf16 multiply (exact product, RNE to bf16) — CDNA5 op 42
__device__ inline unsigned pkmul_bf16(unsigned a, unsigned b) {
  unsigned d;
  asm("v_pk_mul_bf16 %0, %1, %2" : "=v"(d) : "v"(a), "v"(b));
  return d;
}
__device__ inline float wsum32(float v) {
#pragma unroll
  for (int o = 16; o > 0; o >>= 1) v += __shfl_xor(v, o, 32);
  return v;
}
__device__ inline float ln_relu(float v, float g, float b) {
  float mu  = wsum32(v) * 0.03125f;
  float d   = v - mu;
  float var = wsum32(d * d) * 0.03125f;
  float y   = d * rsqrtf(var + 1e-5f) * g + b;
  return fmaxf(y, 0.0f);
}
// 16-bit A/B fragment K-offset pattern (ISA 7.12.2): lane group g in {0,1},
// VGPR j in 0..7 holds K pair { koff, koff+1 }.
__device__ inline int koff(int j, int g) {
  return ((j < 4) ? (2 * j) : (2 * j + 8)) + 8 * g;
}

// ---------------------------------------------------------------------------
// Kernel 1: per-token hypernet (both branches). One wave32 per token,
// lane == hidden unit (HID == 32 == wave width). Writes h2w[t,32], bias[t,128].
// ---------------------------------------------------------------------------
__global__ __launch_bounds__(256) void hyper_kernel(
    const float* __restrict__ z,
    const float* __restrict__ wW1, const float* __restrict__ wb1,
    const float* __restrict__ wg1, const float* __restrict__ wbe1,
    const float* __restrict__ wW2, const float* __restrict__ wb2,
    const float* __restrict__ wg2, const float* __restrict__ wbe2,
    const float* __restrict__ bW1, const float* __restrict__ bb1,
    const float* __restrict__ bg1, const float* __restrict__ bbe1,
    const float* __restrict__ bW2, const float* __restrict__ bb2,
    const float* __restrict__ bg2, const float* __restrict__ bbe2,
    const float* __restrict__ bW3, const float* __restrict__ bb3,
    float* __restrict__ h2out, float* __restrict__ biasout) {
  int gt   = blockIdx.x * blockDim.x + threadIdx.x;
  int t    = gt >> 5;
  int lane = gt & 31;
  if (t >= T_TOK) return;
  float z0 = z[t * 3 + 0], z1 = z[t * 3 + 1], z2 = z[t * 3 + 2];

  // ---- weight branch: 3 -> 32 -> 32 (LN+ReLU twice) ----
  float h = wW1[lane * 3 + 0] * z0 + wW1[lane * 3 + 1] * z1 +
            wW1[lane * 3 + 2] * z2 + wb1[lane];
  h = ln_relu(h, wg1[lane], wbe1[lane]);
  float h2 = wb2[lane];
#pragma unroll
  for (int k = 0; k < 32; k++) h2 = fmaf(wW2[lane * 32 + k], __shfl(h, k, 32), h2);
  h2 = ln_relu(h2, wg2[lane], wbe2[lane]);
  h2out[t * 32 + lane] = h2;

  // ---- bias branch: 3 -> 32 -> 32 -> 128 ----
  float p = bW1[lane * 3 + 0] * z0 + bW1[lane * 3 + 1] * z1 +
            bW1[lane * 3 + 2] * z2 + bb1[lane];
  p = ln_relu(p, bg1[lane], bbe1[lane]);
  float q = bb2[lane];
#pragma unroll
  for (int k = 0; k < 32; k++) q = fmaf(bW2[lane * 32 + k], __shfl(p, k, 32), q);
  q = ln_relu(q, bg2[lane], bbe2[lane]);
#pragma unroll
  for (int j = 0; j < 4; j++) {
    int   o   = j * 32 + lane;
    float acc = bb3[o];
#pragma unroll
    for (int k = 0; k < 32; k++) acc = fmaf(bW3[o * 32 + k], __shfl(q, k, 32), acc);
    biasout[t * 128 + o] = acc;
  }
}

// ---------------------------------------------------------------------------
// Kernel 2: pack wW3 (and wb3 for the 4 K-extension chunks) into per-lane
// bf16 B fragments: Bpack[((c*8+nt)*32+lane)*8 + j] holds B[k0+koff(j,g)..+1][n]
// so the GEMM loads each fragment as two contiguous global_load_b128.
// Grid: 132 blocks (one per chunk) x 256 threads (8 ntiles x 32 lanes).
// ---------------------------------------------------------------------------
__global__ __launch_bounds__(256) void pack_kernel(
    const float* __restrict__ wW3, const float* __restrict__ wb3,
    unsigned* __restrict__ Bpack) {
  int c    = blockIdx.x;         // K chunk 0..131
  int nt   = threadIdx.x >> 5;   // n tile 0..7
  int lane = threadIdx.x & 31;
  int n    = nt * 16 + (lane & 15);
  int g    = lane >> 4;
  unsigned* dst = Bpack + (((c * 8 + nt) * 32 + lane) * 8);
#pragma unroll
  for (int j = 0; j < 8; j++) {
    int   k = koff(j, g);
    float f0, f1;
    if (c < KCH) {               // B[i*32+h][n] = wW3[(n*128+i)*32+h]
      int base = (n * 128 + c) * 32 + k;
      f0 = wW3[base];
      f1 = wW3[base + 1];
    } else {                     // K extension: B[4096+i'][n] = wb3[n*128+i']
      int ip = (c - KCH) * 32 + k;
      f0 = wb3[n * 128 + ip];
      f1 = wb3[n * 128 + ip + 1];
    }
    dst[j] = packbf(f0, f1);
  }
}

// ---------------------------------------------------------------------------
// Kernel 3: GEMM. Each wave owns 32 tokens x 64 out-channels:
// 2 token tiles x 4 n tiles -> 8 f32 WMMA accumulators (init = bias fragment).
// Per 32-K step the A fragment is h2pattern * broadcast(x[t,c]) via
// v_pk_mul_bf16; B fragments stream from the L2-resident Bpack. x is loaded
// as float4 once per 4 K-steps. 1024 waves total = 256 blocks x 128 threads.
// ---------------------------------------------------------------------------
__global__ __launch_bounds__(128) void gemm_kernel(
    const float* __restrict__ x, const float* __restrict__ h2w,
    const float* __restrict__ biasb, const unsigned* __restrict__ Bpack,
    float* __restrict__ out) {
  int w    = (blockIdx.x * 128 + threadIdx.x) >> 5;
  int lane = threadIdx.x & 31;
  if (w >= 1024) return;
  int nhalf = w & 1;             // which half of the 128 out channels
  int t0    = (w >> 1) * 32;     // 32-token super-tile
  int lm    = lane & 15;
  int g     = lane >> 4;
  int tA    = t0 + lm;           // token for this lane, tile 0
  int tB    = tA + 16;           // token for this lane, tile 1

  // per-lane h2 pattern (constant over all 128 main K steps)
  unsigned patA[8], patB[8];
#pragma unroll
  for (int j = 0; j < 8; j++) {
    int h   = koff(j, g);
    patA[j] = packbf(h2w[tA * 32 + h], h2w[tA * 32 + h + 1]);
    patB[j] = packbf(h2w[tB * 32 + h], h2w[tB * 32 + h + 1]);
  }

  // C init = bias fragment (C layout: VGPR r -> M = r + 8*g, lane%16 -> N)
  v8f acc[2][4];
#pragma unroll
  for (int ti = 0; ti < 2; ti++)
#pragma unroll
    for (int nj = 0; nj < 4; nj++)
#pragma unroll
      for (int r = 0; r < 8; r++) {
        int t = t0 + ti * 16 + r + 8 * g;
        int n = (nhalf * 4 + nj) * 16 + lm;
        acc[ti][nj][r] = biasb[t * 128 + n];
      }

  // ---- main K loop: 4 chunks (input channels) per iteration ----
  for (int c4 = 0; c4 < KCH / 4; c4++) {
    float4 xA4 = *(const float4*)(x + tA * 128 + c4 * 4);
    float4 xB4 = *(const float4*)(x + tB * 128 + c4 * 4);
    float  xa[4] = {xA4.x, xA4.y, xA4.z, xA4.w};
    float  xb[4] = {xB4.x, xB4.y, xB4.z, xB4.w};
#pragma unroll
    for (int u = 0; u < 4; u++) {
      int      c   = c4 * 4 + u;
      unsigned xpA = 0x00010001u * (unsigned)f2bf(xa[u]);
      unsigned xpB = 0x00010001u * (unsigned)f2bf(xb[u]);
      v8u aAu, aBu;
#pragma unroll
      for (int j = 0; j < 8; j++) {
        aAu[j] = pkmul_bf16(patA[j], xpA);
        aBu[j] = pkmul_bf16(patB[j], xpB);
      }
      v16bf fa = __builtin_bit_cast(v16bf, aAu);
      v16bf fb = __builtin_bit_cast(v16bf, aBu);
#pragma unroll
      for (int nj = 0; nj < 4; nj++) {
        v8u   bu = *(const v8u*)(Bpack + (((c * 8 + (nhalf * 4 + nj)) * 32 + lane) * 8));
        v16bf bf = __builtin_bit_cast(v16bf, bu);
        acc[0][nj] = __builtin_amdgcn_wmma_f32_16x16x32_bf16(
            false, fa, false, bf, (short)0, acc[0][nj], false, false);
        acc[1][nj] = __builtin_amdgcn_wmma_f32_16x16x32_bf16(
            false, fb, false, bf, (short)0, acc[1][nj], false, false);
      }
    }
  }

  // ---- K extension: fold wb3 (A = x directly), chunks 128..131 ----
#pragma unroll
  for (int cc = KCH; cc < KCH_TOT; cc++) {
    v8u aAu, aBu;
#pragma unroll
    for (int j = 0; j < 8; j++) {
      int ip = (cc - KCH) * 32 + koff(j, g);
      aAu[j] = packbf(x[tA * 128 + ip], x[tA * 128 + ip + 1]);
      aBu[j] = packbf(x[tB * 128 + ip], x[tB * 128 + ip + 1]);
    }
    v16bf fa = __builtin_bit_cast(v16bf, aAu);
    v16bf fb = __builtin_bit_cast(v16bf, aBu);
#pragma unroll
    for (int nj = 0; nj < 4; nj++) {
      v8u   bu = *(const v8u*)(Bpack + (((cc * 8 + (nhalf * 4 + nj)) * 32 + lane) * 8));
      v16bf bf = __builtin_bit_cast(v16bf, bu);
      acc[0][nj] = __builtin_amdgcn_wmma_f32_16x16x32_bf16(
          false, fa, false, bf, (short)0, acc[0][nj], false, false);
      acc[1][nj] = __builtin_amdgcn_wmma_f32_16x16x32_bf16(
          false, fb, false, bf, (short)0, acc[1][nj], false, false);
    }
  }

  // ---- store ----
#pragma unroll
  for (int ti = 0; ti < 2; ti++)
#pragma unroll
    for (int nj = 0; nj < 4; nj++)
#pragma unroll
      for (int r = 0; r < 8; r++) {
        int t = t0 + ti * 16 + r + 8 * g;
        int n = (nhalf * 4 + nj) * 16 + lm;
        out[t * 128 + n] = acc[ti][nj][r];
      }
}

// ---------------------------------------------------------------------------
extern "C" void kernel_launch(void* const* d_in, const int* in_sizes, int n_in,
                              void* d_out, int out_size, void* d_ws, size_t ws_size,
                              hipStream_t stream) {
  const float* x    = (const float*)d_in[0];
  const float* z    = (const float*)d_in[1];
  const float* wW1  = (const float*)d_in[2];
  const float* wb1  = (const float*)d_in[3];
  const float* wg1  = (const float*)d_in[4];
  const float* wbe1 = (const float*)d_in[5];
  const float* wW2  = (const float*)d_in[6];
  const float* wb2  = (const float*)d_in[7];
  const float* wg2  = (const float*)d_in[8];
  const float* wbe2 = (const float*)d_in[9];
  const float* wW3  = (const float*)d_in[10];
  const float* wb3  = (const float*)d_in[11];
  const float* bW1  = (const float*)d_in[12];
  const float* bb1  = (const float*)d_in[13];
  const float* bg1  = (const float*)d_in[14];
  const float* bbe1 = (const float*)d_in[15];
  const float* bW2  = (const float*)d_in[16];
  const float* bb2  = (const float*)d_in[17];
  const float* bg2  = (const float*)d_in[18];
  const float* bbe2 = (const float*)d_in[19];
  const float* bW3  = (const float*)d_in[20];
  const float* bb3  = (const float*)d_in[21];

  // workspace layout: Bpack (~1.06 MB) @0, h2 (2 MB) @2MB, bias (8 MB) @4MB
  char*     ws    = (char*)d_ws;
  unsigned* Bpack = (unsigned*)ws;
  float*    h2w   = (float*)(ws + (2ull << 20));
  float*    biasb = (float*)(ws + (4ull << 20));

  hyper_kernel<<<T_TOK * 32 / 256, 256, 0, stream>>>(
      z, wW1, wb1, wg1, wbe1, wW2, wb2, wg2, wbe2,
      bW1, bb1, bg1, bbe1, bW2, bb2, bg2, bbe2, bW3, bb3, h2w, biasb);
  pack_kernel<<<KCH_TOT, 256, 0, stream>>>(wW3, wb3, Bpack);
  gemm_kernel<<<256, 128, 0, stream>>>(x, h2w, biasb, Bpack, (float*)d_out);
}
